// CodePredLayersExport_8650064134911
// MI455X (gfx1250) — matrix operand order repaired and checked by reading, not compiled, not run on hardware
//
#include <hip/hip_runtime.h>
#include <hip/hip_bf16.h>
#include <math.h>

// ---------------- CDNA5 WMMA types ----------------
typedef __attribute__((ext_vector_type(16))) _Float16 v16h;
typedef __attribute__((ext_vector_type(8)))  float    v8f;

#define L_     5
#define H_     1024
#define NH_    16
#define NKV_   8
#define HD_    128
#define I_     3072
#define MAXS_  16
#define EPS_   1e-6f
#define KVSZ_  (NKV_*MAXS_*HD_)   // 16384 per layer

// ---------------- helpers ----------------
__device__ __forceinline__ float wave_red_sum(float v) {
#pragma unroll
  for (int o = 16; o; o >>= 1) v += __shfl_xor(v, o, 32);
  return v;
}

__device__ __forceinline__ float dotrow(const float* __restrict__ w,
                                        const float* __restrict__ v,
                                        int C4, int lane) {
  const float4* w4 = (const float4*)w;
  const float4* v4 = (const float4*)v;
  float acc = 0.f;
  for (int i = lane; i < C4; i += 32) {
    float4 a = w4[i], b = v4[i];
    acc += a.x * b.x + a.y * b.y + a.z * b.z + a.w * b.w;
  }
  return wave_red_sum(acc);
}

// ---------------- RMSNorm over 1024, one block of 256 ----------------
__global__ void k_rmsnorm(const float* __restrict__ x, const float* __restrict__ w,
                          float* __restrict__ out) {
  __shared__ float red[8];
  const int tid = threadIdx.x;
  float4 xv = ((const float4*)x)[tid];
  float ss = xv.x*xv.x + xv.y*xv.y + xv.z*xv.z + xv.w*xv.w;
  ss = wave_red_sum(ss);
  if ((tid & 31) == 0) red[tid >> 5] = ss;
  __syncthreads();
  float tot = 0.f;
#pragma unroll
  for (int i = 0; i < 8; ++i) tot += red[i];
  const float inv = rsqrtf(tot / (float)H_ + EPS_);
  float4 wv = ((const float4*)w)[tid];
  float4 o4;
  o4.x = xv.x * inv * wv.x; o4.y = xv.y * inv * wv.y;
  o4.z = xv.z * inv * wv.z; o4.w = xv.w * inv * wv.w;
  ((float4*)out)[tid] = o4;
}

// ---------------- QKV matvec: 4096 rows x 1024, row per wave ----------------
__global__ void k_qkv(const float* __restrict__ wq, const float* __restrict__ wk,
                      const float* __restrict__ wv, const float* __restrict__ h,
                      float* __restrict__ q, float* __restrict__ kb, float* __restrict__ vb) {
  const int lane = threadIdx.x & 31;
  const int row  = blockIdx.x * 8 + (threadIdx.x >> 5);
  const float* w; float* out; int orow;
  if (row < 2048)      { w = wq + (size_t)row * H_;          out = q;  orow = row; }
  else if (row < 3072) { w = wk + (size_t)(row - 2048) * H_; out = kb; orow = row - 2048; }
  else                 { w = wv + (size_t)(row - 3072) * H_; out = vb; orow = row - 3072; }
  float r = dotrow(w, h, H_ / 4, lane);
  if (lane == 0) out[orow] = r;
}

// ---------------- per-head q/k norm + RoPE + cache write ----------------
// grid 25 x 128: blocks 0..15 q heads, 16..23 k heads, 24 v copy
__global__ void k_prep(float* __restrict__ q, const float* __restrict__ kb,
                       const float* __restrict__ vb,
                       const float* __restrict__ wqn, const float* __restrict__ wkn,
                       const int* __restrict__ pos_p,
                       float* __restrict__ kcache, float* __restrict__ vcache) {
  const int b = blockIdx.x, tid = threadIdx.x;
  const int pos = pos_p[0];
  if (b == 24) {
    for (int i = tid; i < NKV_ * HD_; i += 128) {
      int g = i >> 7, d = i & 127;
      vcache[(g * MAXS_ + pos) * HD_ + d] = vb[i];
    }
    return;
  }
  __shared__ float buf[HD_];
  __shared__ float red[4];
  const bool isq = (b < 16);
  const int head = isq ? b : b - 16;
  float v = isq ? q[head * HD_ + tid] : kb[head * HD_ + tid];
  float ss = wave_red_sum(v * v);
  if ((tid & 31) == 0) red[tid >> 5] = ss;
  __syncthreads();
  float tot = red[0] + red[1] + red[2] + red[3];
  float inv = rsqrtf(tot / (float)HD_ + EPS_);
  float nv = v * inv * (isq ? wqn[tid] : wkn[tid]);
  buf[tid] = nv;
  __syncthreads();
  // inv_freq[j] = THETA^(-j/64), j = tid % 64
  float fr = (float)pos * expf(-(float)(tid & 63) * (logf(1000000.0f) / 64.0f));
  float c = cosf(fr), s = sinf(fr);
  float rot = (tid < 64) ? -buf[tid + 64] : buf[tid - 64];
  float o = nv * c + rot * s;
  if (isq) q[head * HD_ + tid] = o;
  else     kcache[(head * MAXS_ + pos) * HD_ + tid] = o;
}

// ---------------- GQA attention via WMMA, 1 block x 256 (8 waves) ----------------
__global__ void k_attn(const float* __restrict__ q, const float* __restrict__ kc,
                       const float* __restrict__ vc, const int* __restrict__ pos_p,
                       float* __restrict__ attn) {
  __shared__ float S[8][16][16];
  __shared__ float P[8][16][16];
  const int tid  = threadIdx.x;
  const int lane = tid & 31;
  const int g    = tid >> 5;            // kv head, one per wave
  const int nrow = lane & 15;           // A-row (head) / B-col index
  const int hi8  = (lane & 16) ? 8  : 0;
  const int lo16 = (lane & 16) ? 16 : 0;
  const int pos  = pos_p[0];

  // ---- scores: S = Q (16x128) @ Kg^T (128x16), fp16 in / fp32 acc ----
  v8f c = {};
#pragma unroll
  for (int chunk = 0; chunk < 4; ++chunk) {
    const int d0 = chunk * 32;
    v16h a, b;
#pragma unroll
    for (int e = 0; e < 16; ++e) {
      int ka = (e < 8) ? (hi8 + e) : (16 + hi8 + (e - 8)); // A f16 16x32 layout
      int kb = (e < 8) ? (hi8 + e) : (16 + hi8 + (e - 8)); // B f16 32x16 layout
      a[e] = (_Float16)q[nrow * HD_ + d0 + ka];
      b[e] = (_Float16)kc[(g * MAXS_ + nrow) * HD_ + d0 + kb];
    }
    c = __builtin_amdgcn_wmma_f32_16x16x32_f16(false, a, false, b, (short)0, c, false, false);
  }
  {
    const int mo = (lane & 16) ? 8 : 0;
#pragma unroll
    for (int j = 0; j < 8; ++j) S[g][mo + j][nrow] = c[j];
  }
  __syncthreads();

  // ---- masked softmax on this wave's two rows (2g lanes0-15, 2g+1 lanes16-31) ----
  {
    const int m = 2 * g + ((lane & 16) ? 1 : 0);
    const int n = nrow;
    const float scale = 0.08838834764831845f; // 1/sqrt(128)
    float sv = (n <= pos) ? S[g][m][n] * scale : -INFINITY;
    float mx = sv;
#pragma unroll
    for (int o = 8; o; o >>= 1) mx = fmaxf(mx, __shfl_xor(mx, o, 16));
    float ev = (n <= pos) ? expf(sv - mx) : 0.f;
    float sum = ev;
#pragma unroll
    for (int o = 8; o; o >>= 1) sum += __shfl_xor(sum, o, 16);
    float p = ev / sum;
    float* pp = &P[g][0][0];
    for (int i = lane; i < 256; i += 32) pp[i] = 0.f;
    __syncthreads();
    P[g][m][n] = p;  // only this wave's 2 rows are non-zero
  }
  __syncthreads();

  // ---- attn = P (16x16, K padded to 32) @ Vg (16x128), 8 column tiles ----
  const int mo = (lane & 16) ? 8 : 0;
#pragma unroll
  for (int t = 0; t < 8; ++t) {
    v16h ap, bv;
#pragma unroll
    for (int e = 0; e < 16; ++e) {
      int ka = (e < 8) ? (hi8 + e) : (16 + hi8 + (e - 8));
      int kb = (e < 8) ? (hi8 + e) : (16 + hi8 + (e - 8));
      ap[e] = (ka < 16) ? (_Float16)P[g][nrow][ka] : (_Float16)0.f;
      bv[e] = (kb < 16) ? (_Float16)vc[(g * MAXS_ + kb) * HD_ + t * 16 + nrow] : (_Float16)0.f;
    }
    v8f c2 = {};
    c2 = __builtin_amdgcn_wmma_f32_16x16x32_f16(false, ap, false, bv, (short)0, c2, false, false);
#pragma unroll
    for (int j = 0; j < 8; ++j) {
      int m = j + mo;
      if (m == 2 * g || m == 2 * g + 1)
        attn[m * HD_ + t * 16 + nrow] = c2[j];
    }
  }
}

// ---------------- O-proj + residual: 1024 rows x 2048 ----------------
__global__ void k_oproj(const float* __restrict__ wo, const float* __restrict__ attn,
                        float* __restrict__ x) {
  const int lane = threadIdx.x & 31;
  const int row  = blockIdx.x * 8 + (threadIdx.x >> 5);
  float r = dotrow(wo + (size_t)row * 2048, attn, 2048 / 4, lane);
  if (lane == 0) x[row] += r;
}

// ---------------- gate/up fused: 3072 rows, m = silu(g)*u ----------------
__global__ void k_gateup(const float* __restrict__ wg, const float* __restrict__ wu,
                         const float* __restrict__ h, float* __restrict__ m) {
  const int lane = threadIdx.x & 31;
  const int row  = blockIdx.x * 8 + (threadIdx.x >> 5);
  float gg = dotrow(wg + (size_t)row * H_, h, H_ / 4, lane);
  float uu = dotrow(wu + (size_t)row * H_, h, H_ / 4, lane);
  if (lane == 0) m[row] = (gg / (1.f + expf(-gg))) * uu;
}

// ---------------- down + residual: 1024 rows x 3072 ----------------
__global__ void k_down(const float* __restrict__ wd, const float* __restrict__ m,
                       float* __restrict__ x) {
  const int lane = threadIdx.x & 31;
  const int row  = blockIdx.x * 8 + (threadIdx.x >> 5);
  float r = dotrow(wd + (size_t)row * I_, m, I_ / 4, lane);
  if (lane == 0) x[row] += r;
}

// ---------------- launch ----------------
extern "C" void kernel_launch(void* const* d_in, const int* in_sizes, int n_in,
                              void* d_out, int out_size, void* d_ws, size_t ws_size,
                              hipStream_t stream) {
  const float* hs    = (const float*)d_in[0];
  const int*   pos   = (const int*)  d_in[1];
  const float* pk    = (const float*)d_in[2];
  const float* pv    = (const float*)d_in[3];
  const float* w_iln = (const float*)d_in[4];
  const float* w_paln= (const float*)d_in[5];
  const float* wq    = (const float*)d_in[6];
  const float* wk    = (const float*)d_in[7];
  const float* wv    = (const float*)d_in[8];
  const float* wo    = (const float*)d_in[9];
  const float* w_qn  = (const float*)d_in[10];
  const float* w_kn  = (const float*)d_in[11];
  const float* wg    = (const float*)d_in[12];
  const float* wu    = (const float*)d_in[13];
  const float* wd    = (const float*)d_in[14];
  const float* w_on  = (const float*)d_in[15];

  float* out   = (float*)d_out;
  float* pks_o = out + H_;
  float* pvs_o = pks_o + L_ * KVSZ_;

  float* ws   = (float*)d_ws;
  float* x    = ws;            // 1024
  float* h    = ws + 1024;     // 1024
  float* q    = ws + 2048;     // 2048
  float* kb   = ws + 4096;     // 1024
  float* vb   = ws + 5120;     // 1024
  float* attn = ws + 6144;     // 2048
  float* mm   = ws + 8192;     // 3072

  hipMemcpyAsync(x,     hs, H_ * sizeof(float),          hipMemcpyDeviceToDevice, stream);
  hipMemcpyAsync(pks_o, pk, L_ * KVSZ_ * sizeof(float),  hipMemcpyDeviceToDevice, stream);
  hipMemcpyAsync(pvs_o, pv, L_ * KVSZ_ * sizeof(float),  hipMemcpyDeviceToDevice, stream);

  for (int l = 0; l < L_; ++l) {
    float* kc = pks_o + (size_t)l * KVSZ_;
    float* vc = pvs_o + (size_t)l * KVSZ_;
    k_rmsnorm<<<1, 256, 0, stream>>>(x, w_iln + l * H_, h);
    k_qkv<<<512, 256, 0, stream>>>(wq + (size_t)l * 2048 * H_,
                                   wk + (size_t)l * 1024 * H_,
                                   wv + (size_t)l * 1024 * H_, h, q, kb, vb);
    k_prep<<<25, 128, 0, stream>>>(q, kb, vb, w_qn + l * HD_, w_kn + l * HD_, pos, kc, vc);
    k_attn<<<1, 256, 0, stream>>>(q, kc, vc, pos, attn);
    k_oproj<<<128, 256, 0, stream>>>(wo + (size_t)l * H_ * 2048, attn, x);
    k_rmsnorm<<<1, 256, 0, stream>>>(x, w_paln + l * H_, h);
    k_gateup<<<384, 256, 0, stream>>>(wg + (size_t)l * I_ * H_,
                                      wu + (size_t)l * I_ * H_, h, mm);
    k_down<<<128, 256, 0, stream>>>(wd + (size_t)l * H_ * I_, mm, x);
  }
  k_rmsnorm<<<1, 256, 0, stream>>>(x, w_on, out);
}
